// SelfAttention_26474178412977
// MI455X (gfx1250) — matrix operand order repaired and checked by reading, compiled, not run on hardware
//
#include <hip/hip_runtime.h>
#include <hip/hip_bf16.h>

typedef __attribute__((ext_vector_type(16))) _Float16 v16h;
typedef __attribute__((ext_vector_type(8)))  _Float16 v8h;
typedef __attribute__((ext_vector_type(8)))  float    v8f;
typedef __attribute__((ext_vector_type(4)))  int      v4i;

#define WMMA16(a, b, c) \
  __builtin_amdgcn_wmma_f32_16x16x32_f16(false, (a), false, (b), (short)0, (c), false, false)

// ---- gfx1250 async global->LDS path (guarded; falls back to sync copy) ----
#if defined(__has_builtin)
#if __has_builtin(__builtin_amdgcn_global_load_async_to_lds_b128)
#define HAVE_ASYNC_LDS 1
#endif
#endif
#ifndef HAVE_ASYNC_LDS
#define HAVE_ASYNC_LDS 0
#endif

typedef __attribute__((address_space(1))) v4i as1_v4i;   // global
typedef __attribute__((address_space(3))) v4i as3_v4i;   // LDS

__device__ __forceinline__ void cp16_g2l(const void* g, void* l) {
#if HAVE_ASYNC_LDS
  __builtin_amdgcn_global_load_async_to_lds_b128((as1_v4i*)const_cast<void*>(g),
                                                 (as3_v4i*)l, 0, 0);
#else
  *reinterpret_cast<v8h*>(l) = *reinterpret_cast<const v8h*>(g);
#endif
}

__device__ __forceinline__ void cp_wait() {
#if HAVE_ASYNC_LDS
#if __has_builtin(__builtin_amdgcn_s_wait_asynccnt)
  __builtin_amdgcn_s_wait_asynccnt(0);
#else
  asm volatile("s_wait_asynccnt 0x0" ::: "memory");
#endif
#endif
}

__device__ __forceinline__ v16h cat16(v8h lo, v8h hi) {
  v16h r;
#pragma unroll
  for (int i = 0; i < 8; ++i) { r[i] = lo[i]; r[i + 8] = hi[i]; }
  return r;
}

// ---------------- fp32 -> fp16 convert ----------------
__global__ void cvt_f32_f16(const float* __restrict__ in, _Float16* __restrict__ out, int n) {
  int i = blockIdx.x * blockDim.x + threadIdx.x;
  if (i < n) out[i] = (_Float16)in[i];
}

// ---------------- generic WMMA GEMM: C[M,N] = A[M,K] * B[K,N] ----------------
// mode 0: outF[row*N+col] = acc*scale + bias[col]        (fp32 out)
// mode 1: outH[((b*4+h)*2048+t)*256+e] = acc*scale (f16), b=row>>11,t=row&2047,h=col>>8,e=col&255
__global__ __launch_bounds__(256) void gemm_f16(
    const _Float16* __restrict__ A, const _Float16* __restrict__ B,
    int M, int N, int K, float scale, const float* __restrict__ bias,
    float* __restrict__ outF, _Float16* __restrict__ outH, int mode) {
  __shared__ _Float16 As[64][32];   // row-major [m][k]
  __shared__ _Float16 Bs[64][32];   // transposed [n][k]
  const int tid  = threadIdx.x;
  const int lane = tid & 31, w = tid >> 5;
  const int lrow = lane & 15, lhi = lane >> 4;
  const int m0 = blockIdx.x * 64;
  const int n0 = blockIdx.y * 64;
  const int msub = w & 3;          // which 16-row subtile
  const int np   = (w >> 2) * 2;   // first of 2 n-subtiles

  v8f acc0 = {}; v8f acc1 = {};

  const int ar  = tid >> 2, akoff = (tid & 3) * 8;   // A tile loader
  const int bk  = tid >> 3, bnoff = (tid & 7) * 8;   // B tile loader

  for (int kb = 0; kb < K; kb += 32) {
    __syncthreads();
    // stage A tile: linear copy -> async data mover path when available
    cp16_g2l(A + (size_t)(m0 + ar) * K + kb + akoff, &As[ar][akoff]);
    // stage B tile transposed: read along N, scatter into Bs[n][k]
    {
      v8h v = *reinterpret_cast<const v8h*>(B + (size_t)(kb + bk) * N + n0 + bnoff);
#pragma unroll
      for (int i = 0; i < 8; ++i) Bs[bnoff + i][bk] = v[i];
    }
    if (kb + 32 < K) {
      __builtin_prefetch(A + (size_t)(m0 + ar) * K + kb + 32 + akoff, 0, 1);
      __builtin_prefetch(B + (size_t)(kb + 32 + bk) * N + n0 + bnoff, 0, 1);
    }
    cp_wait();
    __syncthreads();

    // A fragment: lane holds row = msub*16+lrow; K = 16*lhi..+15 (contiguous)
    v16h af;
    {
      const int r = msub * 16 + lrow, k0 = lhi * 16;
      af = cat16(*reinterpret_cast<const v8h*>(&As[r][k0]),
                 *reinterpret_cast<const v8h*>(&As[r][k0 + 8]));
    }
    // B fragments: lane holds col n = nsub*16+lrow; K = 16*lhi..+15 (contiguous)
    const int bk0 = lhi * 16;
    v16h b0 = cat16(*reinterpret_cast<const v8h*>(&Bs[np * 16 + lrow][bk0]),
                    *reinterpret_cast<const v8h*>(&Bs[np * 16 + lrow][bk0 + 8]));
    v16h b1 = cat16(*reinterpret_cast<const v8h*>(&Bs[(np + 1) * 16 + lrow][bk0]),
                    *reinterpret_cast<const v8h*>(&Bs[(np + 1) * 16 + lrow][bk0 + 8]));
    acc0 = WMMA16(af, b0, acc0);
    acc1 = WMMA16(af, b1, acc1);
  }

  // epilogue: C/D layout: lane l, VGPR j -> (M = j + 8*(l>>4), N = l&15)
#pragma unroll
  for (int p = 0; p < 2; ++p) {
    v8f acc = p ? acc1 : acc0;
    const int col = n0 + (np + p) * 16 + lrow;
#pragma unroll
    for (int j = 0; j < 8; ++j) {
      const int row = m0 + msub * 16 + j + 8 * lhi;
      const float v = acc[j] * scale;
      if (mode == 0) {
        outF[(size_t)row * N + col] = v + bias[col];
      } else {
        const int b = row >> 11, t = row & 2047, hh = col >> 8, e = col & 255;
        outH[(((size_t)b * 4 + hh) * 2048 + t) * 256 + e] = (_Float16)v;
      }
    }
  }
}

// ---------------- flash attention per (b,h): q,k,v are [64][2048][256] f16 ----------------
__global__ __launch_bounds__(256) void attn_f16(
    const _Float16* __restrict__ Q, const _Float16* __restrict__ Km,
    const _Float16* __restrict__ V, _Float16* __restrict__ O) {
  __shared__ _Float16 Ks[32][256];     // [key][e] row-major (B-frag for Q*K^T)
  __shared__ _Float16 Vt[256][32];     // [e][key] transposed (B-frag for P*V)
  __shared__ _Float16 Ps[8][16][32];   // per-wave P tile [m][key]
  const int tid = threadIdx.x, lane = tid & 31, w = tid >> 5;
  const int lrow = lane & 15, lhi = lane >> 4;
  const int bh = blockIdx.y;
  const int t0 = blockIdx.x * 128 + w * 16;
  const size_t base = (size_t)bh * 2048 * 256;

  // Preload Q fragments in registers: 8 chunks of K(e)=32
  v16h qf[8];
  {
    const _Float16* qrow = Q + base + (size_t)(t0 + lrow) * 256;
#pragma unroll
    for (int s = 0; s < 8; ++s) {
      const int k0 = s * 32 + lhi * 16;
      qf[s] = cat16(*reinterpret_cast<const v8h*>(qrow + k0),
                    *reinterpret_cast<const v8h*>(qrow + k0 + 8));
    }
  }

  float mrow[8], lrun[8];
#pragma unroll
  for (int j = 0; j < 8; ++j) { mrow[j] = -1e30f; lrun[j] = 0.f; }
  v8f oacc[16];
#pragma unroll
  for (int e = 0; e < 16; ++e) oacc[e] = (v8f){};

  for (int jb = 0; jb < 64; ++jb) {
    __syncthreads();
    // stage K block: pure linear copy -> async data mover path when available
    {
      const _Float16* kg = Km + base + (size_t)jb * 32 * 256;
      _Float16* ks = &Ks[0][0];
#pragma unroll
      for (int c = 0; c < 4; ++c) {
        const int off = (tid * 4 + c) * 8;
        cp16_g2l(kg + off, ks + off);
      }
    }
    // stage V block transposed (must round-trip VGPRs)
    {
      const int key = tid >> 3, e0 = (tid & 7) * 32;
      const _Float16* vg = V + base + (size_t)(jb * 32 + key) * 256 + e0;
#pragma unroll
      for (int c = 0; c < 4; ++c) {
        v8h vv = *reinterpret_cast<const v8h*>(vg + c * 8);
#pragma unroll
        for (int i = 0; i < 8; ++i) Vt[e0 + c * 8 + i][key] = vv[i];
      }
    }
    cp_wait();
    __syncthreads();

    // S = Q * K^T : two 16x16 tiles (keys 0-15, 16-31), accumulate over e in 8 steps
    v8f s0 = (v8f){}, s1 = (v8f){};
#pragma unroll
    for (int s = 0; s < 8; ++s) {
      const int k0 = s * 32 + lhi * 16;
      v16h b0 = cat16(*reinterpret_cast<const v8h*>(&Ks[lrow][k0]),
                      *reinterpret_cast<const v8h*>(&Ks[lrow][k0 + 8]));
      v16h b1 = cat16(*reinterpret_cast<const v8h*>(&Ks[16 + lrow][k0]),
                      *reinterpret_cast<const v8h*>(&Ks[16 + lrow][k0 + 8]));
      s0 = WMMA16(qf[s], b0, s0);
      s1 = WMMA16(qf[s], b1, s1);
    }

    // Online softmax: lane owns col N=lrow, rows M = j + 8*lhi
    float sf[8];
#pragma unroll
    for (int j = 0; j < 8; ++j) {
      const float a0 = s0[j], a1 = s1[j];
      float mx = fmaxf(a0, a1);
#pragma unroll
      for (int off = 1; off < 16; off <<= 1) mx = fmaxf(mx, __shfl_xor(mx, off, 32));
      const float mnew = fmaxf(mrow[j], mx);
      const float fs = __expf(mrow[j] - mnew);
      const float p0 = __expf(a0 - mnew);
      const float p1 = __expf(a1 - mnew);
      float sum = p0 + p1;
#pragma unroll
      for (int off = 1; off < 16; off <<= 1) sum += __shfl_xor(sum, off, 32);
      lrun[j] = lrun[j] * fs + sum;
      mrow[j] = mnew;
      sf[j] = fs;
      const int m = j + 8 * lhi;
      Ps[w][m][lrow] = (_Float16)p0;
      Ps[w][m][lrow + 16] = (_Float16)p1;
    }
    // rescale running output
#pragma unroll
    for (int e = 0; e < 16; ++e)
#pragma unroll
      for (int j = 0; j < 8; ++j) oacc[e][j] *= sf[j];

    __syncthreads();  // make Ps visible across lanes (and order LDS ops)

    // A fragment of P (16x32), then O += P * V over 16 e-tiles
    v16h pf;
    {
      const int k0 = lhi * 16;
      pf = cat16(*reinterpret_cast<const v8h*>(&Ps[w][lrow][k0]),
                 *reinterpret_cast<const v8h*>(&Ps[w][lrow][k0 + 8]));
    }
#pragma unroll
    for (int e = 0; e < 16; ++e) {
      const int er = e * 16 + lrow, k0 = lhi * 16;
      v16h vf = cat16(*reinterpret_cast<const v8h*>(&Vt[er][k0]),
                      *reinterpret_cast<const v8h*>(&Vt[er][k0 + 8]));
      oacc[e] = WMMA16(pf, vf, oacc[e]);
    }
  }

  // epilogue: normalize and write att-out as (b, t, h*256 + e) f16
  const int b = bh >> 2, h = bh & 3;
#pragma unroll
  for (int j = 0; j < 8; ++j) {
    const float inv = 1.0f / lrun[j];
    const int t = t0 + j + 8 * lhi;
    const size_t rowbase = ((size_t)b * 2048 + t) * 1024 + (size_t)h * 256;
#pragma unroll
    for (int e = 0; e < 16; ++e)
      O[rowbase + e * 16 + lrow] = (_Float16)(oacc[e][j] * inv);
  }
}

// ---------------- host launch ----------------
extern "C" void kernel_launch(void* const* d_in, const int* in_sizes, int n_in,
                              void* d_out, int out_size, void* d_ws, size_t ws_size,
                              hipStream_t stream) {
  const float* x  = (const float*)d_in[0];
  const float* Wq = (const float*)d_in[1];
  const float* Wk = (const float*)d_in[2];
  const float* Wv = (const float*)d_in[3];
  const float* Wu = (const float*)d_in[4];
  const float* bu = (const float*)d_in[5];

  const int B = 16, T = 2048, E = 256, H = 4;
  const int BT = B * T;            // 32768
  const int EH = E * H;            // 1024
  const size_t N_X = (size_t)BT * E;        // 8,388,608
  const size_t N_W = (size_t)E * EH;        // 262,144
  const size_t N_HEAD = (size_t)B * H * T * E;  // 33,554,432

  _Float16* ws = (_Float16*)d_ws;
  size_t o = 0;
  _Float16* xb  = ws + o; o += N_X;
  _Float16* wqb = ws + o; o += N_W;
  _Float16* wkb = ws + o; o += N_W;
  _Float16* wvb = ws + o; o += N_W;
  _Float16* wub = ws + o; o += N_W;
  _Float16* qh  = ws + o; o += N_HEAD;
  _Float16* kh  = ws + o; o += N_HEAD;
  _Float16* vh  = ws + o; o += N_HEAD;
  _Float16* att = ws + o; o += N_HEAD;   // (b, t, h*e) layout

  // fp32 -> fp16 conversions
  cvt_f32_f16<<<(unsigned)((N_X + 255) / 256), 256, 0, stream>>>(x, xb, (int)N_X);
  cvt_f32_f16<<<(unsigned)((N_W + 255) / 256), 256, 0, stream>>>(Wq, wqb, (int)N_W);
  cvt_f32_f16<<<(unsigned)((N_W + 255) / 256), 256, 0, stream>>>(Wk, wkb, (int)N_W);
  cvt_f32_f16<<<(unsigned)((N_W + 255) / 256), 256, 0, stream>>>(Wv, wvb, (int)N_W);
  cvt_f32_f16<<<(unsigned)((N_W + 255) / 256), 256, 0, stream>>>(Wu, wub, (int)N_W);

  // QKV projections (scale 0.25 = 256^-1/4 folded into q and k)
  dim3 gp(BT / 64, EH / 64);   // (512, 16)
  gemm_f16<<<gp, 256, 0, stream>>>(xb, wqb, BT, EH, E, 0.25f, nullptr, nullptr, qh, 1);
  gemm_f16<<<gp, 256, 0, stream>>>(xb, wkb, BT, EH, E, 0.25f, nullptr, nullptr, kh, 1);
  gemm_f16<<<gp, 256, 0, stream>>>(xb, wvb, BT, EH, E, 1.0f,  nullptr, nullptr, vh, 1);

  // attention: grid (t-tiles of 128, b*h)
  attn_f16<<<dim3(T / 128, B * H), 256, 0, stream>>>(qh, kh, vh, att);

  // output projection + bias (fp32 out)
  dim3 go(BT / 64, E / 64);    // (512, 4)
  gemm_f16<<<go, 256, 0, stream>>>(att, wub, BT, E, EH, 1.0f, bu, (float*)d_out, nullptr, 0);
}
